// Network_6399501271737
// MI455X (gfx1250) — matrix operand order repaired and checked by reading, not compiled, not run on hardware
//
#include <hip/hip_runtime.h>
#include <hip/hip_bf16.h>
#include <math.h>

typedef __bf16 bf16_t;
typedef __attribute__((ext_vector_type(16))) __bf16 v16bf;
typedef __attribute__((ext_vector_type(8)))  __bf16 v8bf;
typedef __attribute__((ext_vector_type(8)))  float  v8f;

#define NTOK 1024
#define DV   400
#define KXP  416      // DV padded to multiple of 32
#define G4   1600     // 4*DV gates
#define KSR  800      // 2*DV (divisible by 32)
#define HH   512
#define FF   128

#define NSPLIT 16             // workgroups per LSTM direction
#define CELLS  (DV / NSPLIT)  // 25 cells per workgroup
#define GROWS  (4 * CELLS)    // 100 gate rows per workgroup

__device__ __forceinline__ float sigf(float x) { return 1.0f / (1.0f + __expf(-x)); }

// ---------------------------------------------------------------------------
// Embedding gather + concat -> bf16, K padded 400->416 with zeros
// ---------------------------------------------------------------------------
__global__ void k_embed(const int* __restrict__ wid, const int* __restrict__ pid,
                        const float* __restrict__ wemb, const float* __restrict__ pemb,
                        bf16_t* __restrict__ x)
{
    int idx = blockIdx.x * blockDim.x + threadIdx.x;
    if (idx >= NTOK * KXP) return;
    int i = idx / KXP, d = idx - i * KXP;
    float v = 0.0f;
    if (d < 300)      v = wemb[(size_t)wid[i] * 300 + d];
    else if (d < 400) v = pemb[(size_t)pid[i] * 100 + (d - 300)];
    x[idx] = (bf16_t)v;
}

// W [N,K] fp32 row-major  ->  WT [Kpad,N] bf16 (zero-padded K rows)
__global__ void k_transconv(const float* __restrict__ W, bf16_t* __restrict__ WT,
                            int N, int K, int Kpad)
{
    int idx = blockIdx.x * blockDim.x + threadIdx.x;
    if (idx >= Kpad * N) return;
    int k = idx / N, n = idx - k * N;
    WT[idx] = (bf16_t)((k < K) ? W[(size_t)n * K + k] : 0.0f);
}

// elementwise fp32 -> bf16 (U_1 is already [K,N] for Hh @ U_1)
__global__ void k_convert(const float* __restrict__ W, bf16_t* __restrict__ WT, int n)
{
    int idx = blockIdx.x * blockDim.x + threadIdx.x;
    if (idx < n) WT[idx] = (bf16_t)W[idx];
}

__global__ void k_biassum(const float* __restrict__ a, const float* __restrict__ b,
                          float* __restrict__ o, int n)
{
    int idx = blockIdx.x * blockDim.x + threadIdx.x;
    if (idx < n) o[idx] = a[idx] + b[idx];
}

// s[i] = dot(Hh[i,:], u2)
__global__ void k_rowdot(const float* __restrict__ Hh, const float* __restrict__ u2,
                         float* __restrict__ s)
{
    int i = blockIdx.x * blockDim.x + threadIdx.x;
    if (i >= NTOK) return;
    float acc = 0.0f;
    for (int k = 0; k < FF; ++k) acc += Hh[(size_t)i * FF + k] * u2[k];
    s[i] = acc;
}

// ---------------------------------------------------------------------------
// Generic WMMA GEMM: C[M,N] = A[M,K](bf16) x BT[K,N](bf16)  (+bias[n]) (+rowAdd[m])
// optional relu; optional f32 store, optional bf16 store (optionally transposed).
// Block: 256 threads = 8 waves; each wave owns a (16*MR)x16 tile: MR f32
// accumulators share one B fragment per K-step -> MR WMMAs per B load.
// A-frag (ISA 16-bit A layout): lane<16 holds K 0-7,16-23 of row M=lane;
// lane>=16 holds K 8-15,24-31 of row M=lane-16 -> two b128 loads / lane.
// B-frag: lane = N column n0+l%16, K k0+koff+{0-7,16-23} -> 16 strided 2B loads / lane.
// C layout: vgpr r, lane l<16 -> (m0+r, n0+l); l>=16 -> (m0+r+8, n0+l-16).
// ---------------------------------------------------------------------------
template<int MR>
__global__ __launch_bounds__(256)
void k_wmma_gemm(const bf16_t* __restrict__ A, const bf16_t* __restrict__ BT,
                 const float* __restrict__ bias, float* __restrict__ outF,
                 bf16_t* __restrict__ outB, const float* __restrict__ rowAdd,
                 int M, int N, int K, int relu, int transB)
{
    const int lane = threadIdx.x & 31;
    const int wave = threadIdx.x >> 5;
    const int m0 = blockIdx.x * (128 * MR) + wave * (16 * MR);
    const int n0 = blockIdx.y * 16;
    if (m0 >= M) return;

    const int lr   = lane & 15;
    const int koff = (lane >> 4) * 8;

    const bf16_t* arow = A  + (size_t)(m0 + lr) * K + koff;
    const bf16_t* brow = BT + (size_t)koff * N + n0 + lr;

    v8f acc[MR];
#pragma unroll
    for (int r = 0; r < MR; ++r) acc[r] = (v8f){};

    for (int k0 = 0; k0 < K; k0 += 32) {
        // prefetch next B row for this lane (global_prefetch_b8)
        if (k0 + 32 < K)
            __builtin_prefetch((const void*)(brow + (size_t)(k0 + 32) * N), 0, 1);

        v16bf b;  // B[k0+koff+{0..7,16..23}, n0+lr]
#pragma unroll
        for (int e = 0; e < 8; ++e) { b[e] = brow[(size_t)(k0 + e) * N]; b[e + 8] = brow[(size_t)(k0 + 16 + e) * N]; }

#pragma unroll
        for (int r = 0; r < MR; ++r) {
            const bf16_t* ap = arow + (size_t)(r * 16) * K + k0;
            v8bf alo = *(const v8bf*)(ap);        // K = k0+koff    .. +7
            v8bf ahi = *(const v8bf*)(ap + 16);   // K = k0+16+koff .. +7
            v16bf a;
#pragma unroll
            for (int e = 0; e < 8; ++e) { a[e] = alo[e]; a[e + 8] = ahi[e]; }
            acc[r] = __builtin_amdgcn_wmma_f32_16x16x32_bf16(
                false, a, false, b, (short)0, acc[r], false, false);
        }
    }

    const int col  = n0 + lr;
    const float ba = bias ? bias[col] : 0.0f;
#pragma unroll
    for (int r = 0; r < MR; ++r) {
#pragma unroll
        for (int rr = 0; rr < 8; ++rr) {
            int row = m0 + r * 16 + koff + rr;
            float v = acc[r][rr] + ba;
            if (relu) v = fmaxf(v, 0.0f);
            if (rowAdd) v += rowAdd[row];
            if (outF) outF[(size_t)row * N + col] = v;
            if (outB) {
                bf16_t bv = (bf16_t)v;
                if (transB) outB[(size_t)col * M + row] = bv;
                else        outB[(size_t)row * N + col] = bv;
            }
        }
    }
}

// ---------------------------------------------------------------------------
// LSTM init: publish h0 into parity-0 h buffer, zero the per-step arrival
// counters (re-run every launch -> deterministic across graph replays).
// ---------------------------------------------------------------------------
__global__ void k_lstm_init(const float* __restrict__ h0, float* __restrict__ hbuf,
                            int* __restrict__ cnt)
{
    int idx = blockIdx.x * blockDim.x + threadIdx.x;
    if (idx < 2 * DV) {
        int dir = idx / DV, j = idx - dir * DV;
        hbuf[dir * 2 * DV + j] = h0[idx];           // parity-0 slot
    }
    if (idx < 2 * NTOK) cnt[idx] = 0;
}

// ---------------------------------------------------------------------------
// Sequential BiLSTM recurrence, split across NSPLIT=16 workgroups per
// direction (grid 16x2, 256 threads = 8 wave32 waves each). Each workgroup
// owns 25 cells; its 100 Whh rows live in LDS as bf16 (80 KB) so the only
// per-step global traffic is the 1.6 KB h-vector exchange, pulled into LDS
// via CDNA5 async global->LDS loads (ASYNCcnt) when available. Cross-
// workgroup step ordering: release/acquire arrival counters + s_sleep spin.
// ---------------------------------------------------------------------------
__global__ __launch_bounds__(256)
void k_lstm_mb(const float* __restrict__ Gf, const float* __restrict__ Gb,
               const float* __restrict__ WhhF, const float* __restrict__ WhhB,
               const float* __restrict__ c0,
               float* __restrict__ hbuf,   // [2][2][DV]
               int* __restrict__ cnt,      // [2][NTOK]
               bf16_t* __restrict__ sr)
{
    const int part = blockIdx.x;
    const int dir  = blockIdx.y;
    const int tid  = threadIdx.x;
    const int lane = tid & 31;
    const int wave = tid >> 5;
    const int base = part * CELLS;

    const float* __restrict__ G   = dir ? Gb   : Gf;
    const float* __restrict__ Whh = dir ? WhhB : WhhF;
    float* __restrict__ hb = hbuf + dir * 2 * DV;
    int*   __restrict__ cn = cnt + dir * NTOK;

    __shared__ __align__(16) bf16_t whh_s[GROWS * DV];   // 80 KB of 320 KB WGP LDS
    __shared__ __align__(16) float  h_s[DV];
    __shared__ float gat_s[GROWS];
    __shared__ float c_s[CELLS];

    // stage this workgroup's 100 Whh rows into LDS (bf16)
    for (int idx = tid; idx < GROWS * DV; idx += 256) {
        int r = idx / DV, col = idx - r * DV;
        int sec = r / CELLS, cell = r - sec * CELLS;
        whh_s[idx] = (bf16_t)Whh[(size_t)(sec * DV + base + cell) * DV + col];
    }
    if (tid < CELLS) c_s[tid] = c0[dir * DV + base + tid];
    __syncthreads();

    for (int t = 0; t < NTOK; ++t) {
        const int par = t & 1;
        const int row = dir ? (NTOK - 1 - t) : t;

        // wait until every workgroup published h(t-1)  (t==0: init kernel did)
        if (t > 0) {
            if (tid == 0) {
                while (__hip_atomic_load(&cn[t - 1], __ATOMIC_ACQUIRE,
                                         __HIP_MEMORY_SCOPE_AGENT) < NSPLIT)
                    __builtin_amdgcn_s_sleep(1);
            }
            __syncthreads();
        }

        // broadcast h(t-1) from L2 into LDS
        const float* __restrict__ hprev = hb + par * DV;
#if __has_builtin(__builtin_amdgcn_global_load_async_to_lds_b32)
        for (int j = tid; j < DV; j += 256)
            __builtin_amdgcn_global_load_async_to_lds_b32(
                (__attribute__((address_space(1))) int*)(hprev + j),
                (__attribute__((address_space(3))) int*)(&h_s[j]), 0, 0);
        asm volatile("s_wait_asynccnt 0x0" ::: "memory");
#else
        for (int j = tid; j < DV; j += 256) h_s[j] = hprev[j];
#endif
        __syncthreads();

        // 100 gate dots of length 400: one gate per wave, lanes strided,
        // cross-lane reduce via shfl_xor (wave32)
        const float* __restrict__ Grow = G + (size_t)row * G4;
        for (int r = wave; r < GROWS; r += 8) {
            const bf16_t* __restrict__ wrow = whh_s + r * DV;
            float acc = 0.0f;
            for (int col = lane; col < DV; col += 32)
                acc += (float)wrow[col] * h_s[col];
#pragma unroll
            for (int m = 16; m > 0; m >>= 1) acc += __shfl_xor(acc, m, 32);
            if (lane == 0) {
                int sec = r / CELLS, cell = r - sec * CELLS;
                gat_s[r] = acc + Grow[sec * DV + base + cell];
            }
        }
        __syncthreads();

        // cell update for this workgroup's 25 cells; publish h(t), emit sr
        if (tid < CELLS) {
            float iv = sigf(gat_s[tid]);
            float fv = sigf(gat_s[CELLS + tid]);
            float gv = tanhf(gat_s[2 * CELLS + tid]);
            float ov = sigf(gat_s[3 * CELLS + tid]);
            float cv = fv * c_s[tid] + iv * gv;
            c_s[tid] = cv;
            float hv = ov * tanhf(cv);
            hb[(par ^ 1) * DV + base + tid] = hv;
            sr[(size_t)row * KSR + dir * DV + base + tid] = (bf16_t)sigf(hv);
        }
        __syncthreads();
        __threadfence();
        if (tid == 0)
            __hip_atomic_fetch_add(&cn[t], 1, __ATOMIC_RELEASE,
                                   __HIP_MEMORY_SCOPE_AGENT);
    }
}

// ---------------------------------------------------------------------------
extern "C" void kernel_launch(void* const* d_in, const int* in_sizes, int n_in,
                              void* d_out, int out_size, void* d_ws, size_t ws_size,
                              hipStream_t stream)
{
    const int*   word_ids = (const int*)d_in[0];
    const int*   pos_ids  = (const int*)d_in[1];
    const float* h0   = (const float*)d_in[2];
    const float* c0   = (const float*)d_in[3];
    const float* wemb = (const float*)d_in[4];
    const float* pemb = (const float*)d_in[5];
    const float* Wih_f = (const float*)d_in[6];
    const float* Whh_f = (const float*)d_in[7];
    const float* bih_f = (const float*)d_in[8];
    const float* bhh_f = (const float*)d_in[9];
    const float* Wih_b = (const float*)d_in[10];
    const float* Whh_b = (const float*)d_in[11];
    const float* bih_b = (const float*)d_in[12];
    const float* bhh_b = (const float*)d_in[13];
    const float* Wh1 = (const float*)d_in[14];
    const float* bh1 = (const float*)d_in[15];
    const float* Wh2 = (const float*)d_in[16];
    const float* bh2 = (const float*)d_in[17];
    const float* Wd1 = (const float*)d_in[18];
    const float* bd1 = (const float*)d_in[19];
    const float* Wd2 = (const float*)d_in[20];
    const float* bd2 = (const float*)d_in[21];
    const float* U_1 = (const float*)d_in[22];
    const float* u_2 = (const float*)d_in[23];

    char* ws = (char*)d_ws;
    size_t off = 0;
    auto walloc = [&](size_t bytes) -> void* {
        off = (off + 255) & ~(size_t)255;
        void* p = ws + off;
        off += bytes;
        return p;
    };

    bf16_t* xbf   = (bf16_t*)walloc((size_t)NTOK * KXP * 2);
    bf16_t* WTf   = (bf16_t*)walloc((size_t)KXP * G4 * 2);
    bf16_t* WTb   = (bf16_t*)walloc((size_t)KXP * G4 * 2);
    float*  biasf = (float*) walloc((size_t)G4 * 4);
    float*  biasb = (float*) walloc((size_t)G4 * 4);
    float*  Gfm   = (float*) walloc((size_t)NTOK * G4 * 4);
    float*  Gbm   = (float*) walloc((size_t)NTOK * G4 * 4);
    bf16_t* srb   = (bf16_t*)walloc((size_t)NTOK * KSR * 2);
    bf16_t* W1hT  = (bf16_t*)walloc((size_t)KSR * HH * 2);
    bf16_t* W1dT  = (bf16_t*)walloc((size_t)KSR * HH * 2);
    bf16_t* W2hT  = (bf16_t*)walloc((size_t)HH * FF * 2);
    bf16_t* W2dT  = (bf16_t*)walloc((size_t)HH * FF * 2);
    bf16_t* H1h   = (bf16_t*)walloc((size_t)NTOK * HH * 2);
    bf16_t* H1d   = (bf16_t*)walloc((size_t)NTOK * HH * 2);
    float*  Hhf   = (float*) walloc((size_t)NTOK * FF * 4);
    bf16_t* Hhb   = (bf16_t*)walloc((size_t)NTOK * FF * 2);
    bf16_t* HdT   = (bf16_t*)walloc((size_t)FF * NTOK * 2);
    bf16_t* U1b   = (bf16_t*)walloc((size_t)FF * FF * 2);
    bf16_t* Tb    = (bf16_t*)walloc((size_t)NTOK * FF * 2);
    float*  svec  = (float*) walloc((size_t)NTOK * 4);
    float*  hbuf  = (float*) walloc((size_t)2 * 2 * DV * 4);
    int*    cntb  = (int*)   walloc((size_t)2 * NTOK * 4);
    (void)ws_size; (void)in_sizes; (void)n_in; (void)out_size;

    auto cdiv = [](int a, int b) { return (a + b - 1) / b; };

    // ---- prep: embeddings, weight transpose+convert, bias sums -------------
    k_embed<<<cdiv(NTOK * KXP, 256), 256, 0, stream>>>(word_ids, pos_ids, wemb, pemb, xbf);
    k_transconv<<<cdiv(KXP * G4, 256), 256, 0, stream>>>(Wih_f, WTf, G4, DV, KXP);
    k_transconv<<<cdiv(KXP * G4, 256), 256, 0, stream>>>(Wih_b, WTb, G4, DV, KXP);
    k_transconv<<<cdiv(KSR * HH, 256), 256, 0, stream>>>(Wh1, W1hT, HH, KSR, KSR);
    k_transconv<<<cdiv(KSR * HH, 256), 256, 0, stream>>>(Wd1, W1dT, HH, KSR, KSR);
    k_transconv<<<cdiv(HH * FF, 256), 256, 0, stream>>>(Wh2, W2hT, FF, HH, HH);
    k_transconv<<<cdiv(HH * FF, 256), 256, 0, stream>>>(Wd2, W2dT, FF, HH, HH);
    k_convert<<<cdiv(FF * FF, 256), 256, 0, stream>>>(U_1, U1b, FF * FF);
    k_biassum<<<cdiv(G4, 256), 256, 0, stream>>>(bih_f, bhh_f, biasf, G4);
    k_biassum<<<cdiv(G4, 256), 256, 0, stream>>>(bih_b, bhh_b, biasb, G4);
    k_lstm_init<<<cdiv(2 * NTOK, 256), 256, 0, stream>>>(h0, hbuf, cntb);

    // ---- input projections: G = x @ Wih^T + (bih + bhh) --------------------
    {
        dim3 g(NTOK / 256, G4 / 16);
        k_wmma_gemm<2><<<g, 256, 0, stream>>>(xbf, WTf, biasf, Gfm, nullptr, nullptr,
                                              NTOK, G4, KXP, 0, 0);
        k_wmma_gemm<2><<<g, 256, 0, stream>>>(xbf, WTb, biasb, Gbm, nullptr, nullptr,
                                              NTOK, G4, KXP, 0, 0);
    }

    // ---- sequential BiLSTM recurrence -> sr = sigmoid(r) bf16 --------------
    k_lstm_mb<<<dim3(NSPLIT, 2), 256, 0, stream>>>(Gfm, Gbm, Whh_f, Whh_b, c0,
                                                   hbuf, cntb, srb);

    // ---- MLP layer 1 (bias + relu, bf16 out) -------------------------------
    {
        dim3 g(NTOK / 256, HH / 16);
        k_wmma_gemm<2><<<g, 256, 0, stream>>>(srb, W1hT, bh1, nullptr, H1h, nullptr,
                                              NTOK, HH, KSR, 1, 0);
        k_wmma_gemm<2><<<g, 256, 0, stream>>>(srb, W1dT, bd1, nullptr, H1d, nullptr,
                                              NTOK, HH, KSR, 1, 0);
    }

    // ---- MLP layer 2: Hh (f32 + bf16), Hd (bf16 stored transposed) ---------
    {
        dim3 g(NTOK / 256, FF / 16);
        k_wmma_gemm<2><<<g, 256, 0, stream>>>(H1h, W2hT, bh2, Hhf, Hhb, nullptr,
                                              NTOK, FF, HH, 0, 0);
        k_wmma_gemm<2><<<g, 256, 0, stream>>>(H1d, W2dT, bd2, nullptr, HdT, nullptr,
                                              NTOK, FF, HH, 0, 1);
    }

    // ---- s = Hh @ u_2^T ----------------------------------------------------
    k_rowdot<<<cdiv(NTOK, 256), 256, 0, stream>>>(Hhf, u_2, svec);

    // ---- T = Hh @ U_1 ------------------------------------------------------
    {
        dim3 g(NTOK / 256, FF / 16);
        k_wmma_gemm<2><<<g, 256, 0, stream>>>(Hhb, U1b, nullptr, nullptr, Tb, nullptr,
                                              NTOK, FF, FF, 0, 0);
    }

    // ---- adj = T @ Hd^T + s[i]  -> d_out -----------------------------------
    {
        dim3 g(NTOK / 256, NTOK / 16);
        k_wmma_gemm<2><<<g, 256, 0, stream>>>(Tb, HdT, nullptr, (float*)d_out, nullptr,
                                              svec, NTOK, NTOK, FF, 0, 0);
    }
}
